// DCNv2_Circle_85117661872713
// MI455X (gfx1250) — matrix operand order repaired and checked by reading, not compiled, not run on hardware
//
#include <hip/hip_runtime.h>
#include <hip/hip_bf16.h>

typedef __attribute__((ext_vector_type(16))) __bf16 v16bf;
typedef __attribute__((ext_vector_type(8)))  float  v8f;

#define CIN   256
#define COUT  256
#define HW    4096      // 64*64
#define K2    9
#define KTOT  (CIN * K2)        // 2304
#define KB_TOT (KTOT / 32)      // 72 K-chunks of 32
#define KB_STAGE 24             // K-chunks staged in LDS per round
#define NSTAGE (KB_TOT / KB_STAGE)  // 3
#define ROWS_STAGE (KB_STAGE * 32)  // 768
#define PXT   32                // pixel tile per block (two 16-wide N sub-tiles)

__device__ __forceinline__ float blend_circle(const float* w9, int kq) {
    const float A = 0.7071067811865476f;
    const float Bc = 1.0f - A;
    switch (kq) {
        case 0: return A * (A * w9[0] + Bc * w9[1]) + Bc * (A * w9[3] + Bc * w9[4]);
        case 2: return A * (Bc * w9[1] + A * w9[2]) + Bc * (Bc * w9[4] + A * w9[5]);
        case 6: return Bc * (A * w9[3] + Bc * w9[4]) + A * (A * w9[6] + Bc * w9[7]);
        case 8: return Bc * (Bc * w9[4] + A * w9[5]) + A * (Bc * w9[7] + A * w9[8]);
        default: return w9[kq];
    }
}

// Pre-swizzle blended weights into WMMA A-fragment order (bf16).
// Apack element index = ((kb*16 + mt)*32 + lane)*16 + h
//   lane<16 : M = mt*16+lane,    halfs h -> K = kb*32 + (h&7) + (h>>3)*16
//   lane>=16: M = mt*16+lane-16, halfs h -> K = kb*32 + (h&7) + 8 + (h>>3)*16
__global__ void dcn_circle_prep_w(const float* __restrict__ weight,
                                  __bf16* __restrict__ Apack) {
    int gid = blockIdx.x * blockDim.x + threadIdx.x;   // 72*16*32*16 = 589824
    if (gid >= KB_TOT * 16 * 32 * 16) return;
    int h    = gid & 15;
    int lane = (gid >> 4) & 31;
    int mt   = (gid >> 9) & 15;
    int kb   = gid >> 13;
    int m = mt * 16 + (lane & 15);
    int klocal = (h & 7) + ((lane >= 16) ? 8 : 0) + ((h >> 3) << 4);
    int r = kb * 32 + klocal;                 // r = c*9 + kq
    int c = r / K2;
    int kq = r - K2 * c;
    const float* w9 = weight + ((size_t)m * CIN + c) * K2;
    Apack[gid] = (__bf16)blend_circle(w9, kq);
}

struct Tap { int i[4]; float w[4]; };

__global__ void __launch_bounds__(256)
dcn_circle_main(const float* __restrict__ xin,
                const float* __restrict__ offset,
                const float* __restrict__ maskin,
                const __bf16* __restrict__ Apack,
                const float* __restrict__ bias,
                float* __restrict__ out) {
    __shared__ Tap s_tap[K2][PXT];                            // 9 KB
    // B fragments: [kb_l][ni][lane][16 halfs], ni = 16-pixel sub-tile
    __shared__ __align__(32) __bf16 s_buf[KB_STAGE * 2 * 32 * 16]; // 48 KB

    const int t      = threadIdx.x;
    const int b      = blockIdx.x >> 7;        // 128 pixel-tiles per batch
    const int tile   = blockIdx.x & 127;
    const int p_base = tile * PXT;

    // ---- Phase A: tap descriptors (corner idx + mask-folded bilinear weights) ----
    for (int e = t; e < K2 * PXT; e += 256) {
        int k  = e / PXT;          // 0..8
        int px = e % PXT;
        int p  = p_base + px;
        int ho = p >> 6, wo = p & 63;
        float dy = offset[((size_t)(b * 2 * K2 + 2 * k)     * HW) + p];
        float dx = offset[((size_t)(b * 2 * K2 + 2 * k + 1) * HW) + p];
        float mv = maskin[((size_t)(b * K2 + k) * HW) + p];
        float y  = (float)(ho - 1 + k / 3) + dy;
        float x  = (float)(wo - 1 + k % 3) + dx;
        float y0 = floorf(y), x0 = floorf(x);
        float ly = y - y0, lx = x - x0;
        float hy = 1.0f - ly, hx = 1.0f - lx;
        int iy0 = (int)y0, ix0 = (int)x0;
        Tap tp;
        #pragma unroll
        for (int cr = 0; cr < 4; ++cr) {
            int yi = iy0 + (cr >> 1);
            int xi = ix0 + (cr & 1);
            bool ok = (yi >= 0) & (yi < 64) & (xi >= 0) & (xi < 64);
            int yc = min(max(yi, 0), 63), xc = min(max(xi, 0), 63);
            tp.i[cr] = yc * 64 + xc;
            float wv = ((cr >> 1) ? ly : hy) * ((cr & 1) ? lx : hx);
            tp.w[cr] = ok ? wv * mv : 0.0f;
        }
        s_tap[k][px] = tp;
    }

    const int lane = t & 31;
    const int wv   = t >> 5;     // wave id 0..7
    const v16bf* Ap = (const v16bf*)Apack;
    const v16bf* Bp = (const v16bf*)s_buf;

    v8f acc00 = {};   // m-tile wv,   pixels [0,16)
    v8f acc01 = {};   // m-tile wv,   pixels [16,32)
    v8f acc10 = {};   // m-tile wv+8, pixels [0,16)
    v8f acc11 = {};   // m-tile wv+8, pixels [16,32)
    __syncthreads();

    for (int s = 0; s < NSTAGE; ++s) {
        // ---- Phase B: gather + bilinear blend into bf16 B-fragments in LDS ----
        #pragma unroll
        for (int i = 0; i < ROWS_STAGE / 256; ++i) {
            int r_in = t + 256 * i;                 // 0..767
            int r    = s * ROWS_STAGE + r_in;
            int c    = r / K2;
            int kq   = r - K2 * c;
            const float* xb = xin + (((size_t)b * CIN + c) << 12);
            int kb_l = r_in >> 5;
            int kk   = r_in & 31;
            int lane_off = (kk & 8) << 1;           // ((kk/8)&1)*16
            int half     = (kk & 7) | ((kk & 16) >> 1);
            #pragma unroll 4
            for (int px = 0; px < PXT; ++px) {
                int ni  = px >> 4;
                int pxl = px & 15;
                Tap tp = s_tap[kq][px];
                float v = tp.w[0] * xb[tp.i[0]] + tp.w[1] * xb[tp.i[1]]
                        + tp.w[2] * xb[tp.i[2]] + tp.w[3] * xb[tp.i[3]];
                s_buf[(((kb_l * 2 + ni) * 32) + pxl + lane_off) * 16 + half] = (__bf16)v;
            }
        }
        __syncthreads();

        // ---- Phase C: WMMA chain; each A frag feeds 2 wmma, each B frag feeds 2 ----
        int kb0 = s * KB_STAGE;
        #pragma unroll 2
        for (int kbl = 0; kbl < KB_STAGE; ++kbl) {
            int kb = kb0 + kbl;
            v16bf b0 = Bp[(kbl * 2 + 0) * 32 + lane];
            v16bf b1 = Bp[(kbl * 2 + 1) * 32 + lane];
            v16bf a0 = Ap[((size_t)(kb * 16 + wv)     ) * 32 + lane];
            v16bf a1 = Ap[((size_t)(kb * 16 + wv + 8)) * 32 + lane];
            int kbn = (kb + 8 < KB_TOT) ? kb + 8 : kb;
            __builtin_prefetch(&Ap[((size_t)(kbn * 16 + wv)) * 32 + lane], 0, 1);
            acc00 = __builtin_amdgcn_wmma_f32_16x16x32_bf16(
                        false, a0, false, b0, (short)0, acc00, false, false);
            acc01 = __builtin_amdgcn_wmma_f32_16x16x32_bf16(
                        false, a0, false, b1, (short)0, acc01, false, false);
            acc10 = __builtin_amdgcn_wmma_f32_16x16x32_bf16(
                        false, a1, false, b0, (short)0, acc10, false, false);
            acc11 = __builtin_amdgcn_wmma_f32_16x16x32_bf16(
                        false, a1, false, b1, (short)0, acc11, false, false);
        }
        __syncthreads();
    }

    // ---- Store: C-fragment layout (VGPR i -> M = i + 8*(lane>=16), N = lane&15) ----
    int mrow = (lane >> 4) * 8;
    int n    = lane & 15;
    #pragma unroll
    for (int i = 0; i < 8; ++i) {
        int M  = i + mrow;
        int o0 = wv * 16 + M;
        int o1 = (wv + 8) * 16 + M;
        float b0 = bias[o0];
        float b1 = bias[o1];
        out[(((size_t)b * COUT + o0) << 12) + p_base + n]      = acc00[i] + b0;
        out[(((size_t)b * COUT + o0) << 12) + p_base + 16 + n] = acc01[i] + b0;
        out[(((size_t)b * COUT + o1) << 12) + p_base + n]      = acc10[i] + b1;
        out[(((size_t)b * COUT + o1) << 12) + p_base + 16 + n] = acc11[i] + b1;
    }
}

extern "C" void kernel_launch(void* const* d_in, const int* in_sizes, int n_in,
                              void* d_out, int out_size, void* d_ws, size_t ws_size,
                              hipStream_t stream) {
    const float* x      = (const float*)d_in[0];
    const float* offset = (const float*)d_in[1];
    const float* mask   = (const float*)d_in[2];
    const float* weight = (const float*)d_in[3];
    const float* bias   = (const float*)d_in[4];
    float* out          = (float*)d_out;
    __bf16* Apack       = (__bf16*)d_ws;   // 1.125 MB of swizzled bf16 weights

    const int prep_elems = KB_TOT * 16 * 32 * 16;
    dcn_circle_prep_w<<<(prep_elems + 255) / 256, 256, 0, stream>>>(weight, Apack);

    // 4 batches * 128 pixel-tiles of 32
    dcn_circle_main<<<4 * 128, 256, 0, stream>>>(x, offset, mask, Apack, bias, out);
}
